// AugmentedNeuralODE_36799279792353
// MI455X (gfx1250) — matrix operand order repaired and checked by reading, not compiled, not run on hardware
//
#include <hip/hip_runtime.h>
#include <cmath>

typedef __attribute__((ext_vector_type(16))) __bf16 v16bf;
typedef __attribute__((ext_vector_type(8)))  float  v8f;

#define INPUT_DIM_ 64
#define DDIM 128      // augmented state dim
#define HDIM 256      // hidden
#define BATCH_ 4096
#define TSTEPS 50
#define ROWS 16       // batch rows per block
#define YPITCH (DDIM + 8)   // padded bf16 row pitch (LDS bank stagger)
#define HPITCH (HDIM + 8)

// Branch-free fast tanh: tanh(x) = 1 - 2/(exp(2x)+1), via hardware exp2/rcp.
// Output feeds a bf16 WMMA operand, so ~1ulp-f32 accuracy is ample.
__device__ __forceinline__ float fast_tanh(float x) {
  // clamp so exp2 can't overflow; tanh saturates to +-1 well before |x|=9
  float xc = fminf(fmaxf(x, -9.0f), 9.0f);
  float e2 = __builtin_amdgcn_exp2f(xc * 2.8853900817779268f); // exp(2x)
  float r  = __builtin_amdgcn_rcpf(e2 + 1.0f);
  return fmaf(-2.0f, r, 1.0f);
}

// One block integrates a 16-row tile of the batch through all 49 RK4 steps.
// Weights are held in registers as WMMA B-fragments (bf16); RK4 state (y, k1..k3)
// lives in the WMMA C-layout registers of each lane. Only the bf16 copy of the
// current state / hidden activations passes through LDS (A-operand staging).
__global__ __launch_bounds__(256, 1)
void node_rk4(const float* __restrict__ x0, const float* __restrict__ tgrid,
              const float* __restrict__ W1, const float* __restrict__ b1,
              const float* __restrict__ W2, const float* __restrict__ b2,
              float* __restrict__ out)
{
  __shared__ __bf16 sY[ROWS * YPITCH];   // current state, bf16, row-major
  __shared__ __bf16 sH[ROWS * HPITCH];   // hidden activations, bf16, row-major

  const int tid  = threadIdx.x;
  const int w    = tid >> 5;        // wave id 0..7
  const int lane = tid & 31;
  const int l15  = lane & 15;
  const int hi   = lane >> 4;       // half-wave select
  const int row0 = blockIdx.x * ROWS;

  // ---- one-time: gather weight B-fragments into registers (bf16) ----
  // B layout assumption (32x16 tile): lane = column (l15), lanes 0-15 hold
  // K=0..7,16..23 of the tile, lanes 16-31 hold K=8..15,24..31 (matches A).
  v16bf w1f[2][4];            // [local n-tile][k-tile], n-tiles 2w, 2w+1
#pragma unroll
  for (int ntl = 0; ntl < 2; ++ntl) {
    const int n = (2 * w + ntl) * 16 + l15;
#pragma unroll
    for (int kt = 0; kt < 4; ++kt) {
#pragma unroll
      for (int j = 0; j < 16; ++j) {
        const int k = kt * 32 + hi * 8 + (j < 8 ? j : j + 8);
        w1f[ntl][kt][j] = (__bf16)W1[k * HDIM + n];
      }
    }
  }
  v16bf w2f[8];               // [k-tile], n-tile = w
  {
    const int n = w * 16 + l15;
#pragma unroll
    for (int kt = 0; kt < 8; ++kt) {
#pragma unroll
      for (int j = 0; j < 16; ++j) {
        const int k = kt * 32 + hi * 8 + (j < 8 ? j : j + 8);
        w2f[kt][j] = (__bf16)W2[k * DDIM + n];
      }
    }
  }
  const float b1v[2] = { b1[(2 * w + 0) * 16 + l15], b1[(2 * w + 1) * 16 + l15] };
  const float b2v    = b2[w * 16 + l15];

  // ---- init state y0 = [x0 | 0]; this (wave,lane) owns column ncol,
  //      rows m = r + 8*hi  (WMMA f32 C/D layout) ----
  const int ncol = w * 16 + l15;
  float yr[8];
#pragma unroll
  for (int r = 0; r < 8; ++r) {
    const int m = r + 8 * hi;
    float v = (ncol < INPUT_DIM_) ? x0[(row0 + m) * INPUT_DIM_ + ncol] : 0.0f;
    yr[r] = v;
    sY[m * YPITCH + ncol] = (__bf16)v;
    if (ncol < INPUT_DIM_)
      out[(size_t)(row0 + m) * INPUT_DIM_ + ncol] = v;   // sol[0] = y0[:, :64]
  }
  __syncthreads();

  float k1r[8], k2r[8], k3r[8];

#pragma unroll 1
  for (int s = 0; s < TSTEPS - 1; ++s) {
    const float dt = tgrid[s + 1] - tgrid[s];
#pragma unroll
    for (int e = 0; e < 4; ++e) {
      // ---- GEMM1: h = tanh(yt @ W1 + b1), write bf16 to sH ----
#pragma unroll
      for (int ntl = 0; ntl < 2; ++ntl) {
        v8f acc;
#pragma unroll
        for (int r = 0; r < 8; ++r) acc[r] = b1v[ntl];
#pragma unroll
        for (int kt = 0; kt < 4; ++kt) {
          // A fragment (16-bit A 16x32 layout): row = l15; two 16B runs.
          const __bf16* p0 = &sY[l15 * YPITCH + kt * 32 + hi * 8];
          v16bf a;
#pragma unroll
          for (int j = 0; j < 8; ++j) { a[j] = p0[j]; a[8 + j] = p0[16 + j]; }
          acc = __builtin_amdgcn_wmma_f32_16x16x32_bf16(
              false, a, false, w1f[ntl][kt], (short)0, acc, false, false);
        }
        const int n = (2 * w + ntl) * 16 + l15;
#pragma unroll
        for (int r = 0; r < 8; ++r)
          sH[(r + 8 * hi) * HPITCH + n] = (__bf16)fast_tanh(acc[r]);
      }
      __syncthreads();

      // ---- GEMM2: kc = h @ W2 + b2 (stays in registers) ----
      v8f acc2;
#pragma unroll
      for (int r = 0; r < 8; ++r) acc2[r] = b2v;
#pragma unroll
      for (int kt = 0; kt < 8; ++kt) {
        const __bf16* p0 = &sH[l15 * HPITCH + kt * 32 + hi * 8];
        v16bf a;
#pragma unroll
        for (int j = 0; j < 8; ++j) { a[j] = p0[j]; a[8 + j] = p0[16 + j]; }
        acc2 = __builtin_amdgcn_wmma_f32_16x16x32_bf16(
            false, a, false, w2f[kt], (short)0, acc2, false, false);
      }

      // ---- RK4 (3/8 rule) combine in C-register domain ----
#pragma unroll
      for (int r = 0; r < 8; ++r) {
        const int m = r + 8 * hi;
        const float kc = acc2[r];
        float ytn;
        if (e == 0) {                      // k1
          k1r[r] = kc;
          ytn = yr[r] + dt * (1.0f / 3.0f) * kc;
        } else if (e == 1) {               // k2
          k2r[r] = kc;
          ytn = yr[r] + dt * (kc - k1r[r] * (1.0f / 3.0f));
        } else if (e == 2) {               // k3
          k3r[r] = kc;
          ytn = yr[r] + dt * (k1r[r] - k2r[r] + kc);
        } else {                           // k4 -> y_{n+1}
          const float ynew =
              yr[r] + dt * 0.125f * (k1r[r] + 3.0f * (k2r[r] + k3r[r]) + kc);
          yr[r] = ynew;
          ytn = ynew;
          if (ncol < INPUT_DIM_)           // wave-uniform branch (waves 0-3)
            out[((size_t)(s + 1) * BATCH_ + (row0 + m)) * INPUT_DIM_ + ncol] = ynew;
        }
        sY[m * YPITCH + ncol] = (__bf16)ytn;
      }
      __syncthreads();
    }
  }
}

extern "C" void kernel_launch(void* const* d_in, const int* in_sizes, int n_in,
                              void* d_out, int out_size, void* d_ws, size_t ws_size,
                              hipStream_t stream) {
  (void)in_sizes; (void)n_in; (void)d_ws; (void)ws_size; (void)out_size;
  const float* x0 = (const float*)d_in[0];
  const float* t  = (const float*)d_in[1];
  const float* W1 = (const float*)d_in[2];
  const float* b1 = (const float*)d_in[3];
  const float* W2 = (const float*)d_in[4];
  const float* b2 = (const float*)d_in[5];
  float* out = (float*)d_out;
  dim3 grid(BATCH_ / ROWS), block(256);
  hipLaunchKernelGGL(node_rk4, grid, block, 0, stream, x0, t, W1, b1, W2, b2, out);
}
